// HierarchicalMagnificationAttention_63548336111700
// MI455X (gfx1250) — matrix operand order repaired and checked by reading, not compiled, not run on hardware
//
#include <hip/hip_runtime.h>

typedef _Float16 half_t;
typedef __attribute__((ext_vector_type(4)))  _Float16 v4h;
typedef __attribute__((ext_vector_type(8)))  _Float16 v8h;
typedef __attribute__((ext_vector_type(16))) _Float16 v16h;
typedef __attribute__((ext_vector_type(8)))  float    v8f;
typedef __attribute__((ext_vector_type(4)))  int      v4i;

#define Bsz   16384
#define Dm    1024
#define DH    128
#define BD    (16384LL * 1024LL)

// GEMM tiling: block = 128(M) x 128(N), 8 waves as 4(M) x 2(N), 32x64 per wave
#define TM  128
#define TN  128
#define TK  32
#define AST 40    // As row stride (halves): 80B, 16B-aligned rows
#define BST 136   // Bs row stride (halves): 272B, 16B-aligned rows (rows are K)

// --------------------------------------------------------------------------
// gfx1250 async global->LDS staging (guarded so both toolchain passes build)
// --------------------------------------------------------------------------
#if defined(__HIP_DEVICE_COMPILE__) && \
    __has_builtin(__builtin_amdgcn_global_load_async_to_lds_b128)
#define HAVE_ASYNC 1
typedef __attribute__((address_space(1))) v4i* g_v4i_ptr;   // global int4*
typedef __attribute__((address_space(3))) v4i* l_v4i_ptr;   // LDS int4*
#else
#define HAVE_ASYNC 0
#endif

__device__ __forceinline__ void stage_a(const half_t* g, half_t* l)
{
#if HAVE_ASYNC
    // LDS generic address low 32 bits == LDS byte offset (aperture rule),
    // so build the AS pointers via integer casts (always well-formed).
    __builtin_amdgcn_global_load_async_to_lds_b128(
        (g_v4i_ptr)(unsigned long long)(uintptr_t)g,
        (l_v4i_ptr)(unsigned int)(uintptr_t)l,
        0, 0);
#else
    *(v8h*)l = *(const v8h*)g;
#endif
}

__device__ __forceinline__ void wait_async()
{
#if HAVE_ASYNC
#if __has_builtin(__builtin_amdgcn_s_wait_asynccnt)
    __builtin_amdgcn_s_wait_asynccnt(0);
#else
    asm volatile("s_wait_asynccnt 0" ::: "memory");
#endif
#endif
}

// ---------------------------------------------------------------------------
// enh_i = f_i + mag_emb[i]; write f16 copies; hier[0] = enh_0 (fp32) to d_out
// ---------------------------------------------------------------------------
__global__ __launch_bounds__(256) void prep_kernel(
    const float* __restrict__ f0, const float* __restrict__ f1,
    const float* __restrict__ f2, const float* __restrict__ f3,
    const float* __restrict__ mag,
    half_t* __restrict__ e0, half_t* __restrict__ e1,
    half_t* __restrict__ e2, half_t* __restrict__ e3,
    float* __restrict__ hier0)
{
    long long gid = (long long)blockIdx.x * blockDim.x + threadIdx.x;
    int fi = (int)(gid >> 24);              // BD == 2^24
    long long idx = gid & (BD - 1);
    int d = (int)(idx & (Dm - 1));
    const float* fs[4] = {f0, f1, f2, f3};
    half_t* es[4] = {e0, e1, e2, e3};
    float v = fs[fi][idx] + mag[fi * Dm + d];
    es[fi][idx] = (half_t)v;
    if (fi == 0) hier0[idx] = v;
}

__global__ void f32_to_f16_kernel(const float* __restrict__ src,
                                  half_t* __restrict__ dst, int n)
{
    int i = blockIdx.x * blockDim.x + threadIdx.x;
    if (i < n) dst[i] = (half_t)src[i];
}

// ---------------------------------------------------------------------------
// Y[M,Nfull] = X[M,K](f16) * W[N,K]^T(f16) + bias[N], fp32 accumulate (WMMA)
// Double-buffered LDS; A staged via async global->LDS DMA when available.
// ---------------------------------------------------------------------------
template <bool F16OUT>
__global__ __launch_bounds__(256) void gemm_kernel(
    const half_t* __restrict__ X, const half_t* __restrict__ W,
    const float* __restrict__ bias, void* __restrict__ Y,
    int K, int Nfull)
{
    __shared__ half_t As[2][TM * AST];   // As[m][k]
    __shared__ half_t Bs[2][TK * BST];   // Bs[k][n]  (transposed on fill)

    const int tid    = threadIdx.x;
    const int lane   = tid & 31;
    const int lane16 = lane & 15;
    const int laneHi = lane >> 4;
    const int wave   = tid >> 5;
    const int waveM  = (wave & 3) * 32;   // 4 wave-rows of 32
    const int waveN  = (wave >> 2) * 64;  // 2 wave-cols of 64
    const long long blockM = (long long)blockIdx.y * TM;
    const int blockN = blockIdx.x * TN;

    // A-tile: 128x32 halves = 512 8-half chunks -> 2 per thread
    const int aRow = tid >> 2;            // 0..63 (and +64)
    const int aKp  = (tid & 3) * 8;
    // B-tile: 128x32 halves = 512 chunks -> 2 per thread
    const int bN  = tid >> 2;             // 0..63 (and +64)
    const int bKp = (tid & 3) * 8;

    v8f acc[2][4] = {};

    const int nK = K / TK;

    // ---- prologue: stage tile 0 into buffer 0
    stage_a(&X[(blockM + aRow) * K + aKp],      &As[0][aRow * AST + aKp]);
    stage_a(&X[(blockM + aRow + 64) * K + aKp], &As[0][(aRow + 64) * AST + aKp]);
    {
        v8h c0 = *(const v8h*)&W[(long long)(blockN + bN) * K + bKp];
        v8h c1 = *(const v8h*)&W[(long long)(blockN + bN + 64) * K + bKp];
#pragma unroll
        for (int e = 0; e < 8; ++e) Bs[0][(bKp + e) * BST + bN]      = c0[e];
#pragma unroll
        for (int e = 0; e < 8; ++e) Bs[0][(bKp + e) * BST + bN + 64] = c1[e];
    }
    wait_async();
    __syncthreads();

    for (int t = 0; t < nK; ++t) {
        const int cur  = t & 1;
        const int nxt  = cur ^ 1;
        const bool more = (t + 1) < nK;
        v8h nb0 = {}, nb1 = {};

        if (more) {   // issue next tile's fill before computing this one
            const int k0 = (t + 1) * TK;
            stage_a(&X[(blockM + aRow) * K + k0 + aKp],
                    &As[nxt][aRow * AST + aKp]);
            stage_a(&X[(blockM + aRow + 64) * K + k0 + aKp],
                    &As[nxt][(aRow + 64) * AST + aKp]);
            nb0 = *(const v8h*)&W[(long long)(blockN + bN) * K + k0 + bKp];
            nb1 = *(const v8h*)&W[(long long)(blockN + bN + 64) * K + k0 + bKp];
        }

        // A fragment: lane16 = M row; lanes 0-15 hold K {0..7,16..23},
        // lanes 16-31 hold K {8..15,24..31}  (ISA 16-bit A 16x32 layout)
        v16h aF[2], bF[4];
#pragma unroll
        for (int i = 0; i < 2; ++i) {
            const half_t* p = &As[cur][(waveM + i * 16 + lane16) * AST + laneHi * 8];
            v8h lo = *(const v8h*)p;
            v8h hi = *(const v8h*)(p + 16);
            aF[i] = __builtin_shufflevector(lo, hi,
                     0,1,2,3,4,5,6,7,8,9,10,11,12,13,14,15);
        }
        // B fragment: lane16 = N column; lanes 0-15 hold K {0..7,16..23},
        // lanes 16-31 hold K {8..15,24..31}
#pragma unroll
        for (int j = 0; j < 4; ++j) {
            const half_t* p = &Bs[cur][(laneHi * 8) * BST + waveN + j * 16 + lane16];
#pragma unroll
            for (int s = 0; s < 16; ++s)
                bF[j][s] = p[(s < 8 ? s : s + 8) * BST];
        }
#pragma unroll
        for (int i = 0; i < 2; ++i)
#pragma unroll
            for (int j = 0; j < 4; ++j)
                acc[i][j] = __builtin_amdgcn_wmma_f32_16x16x32_f16(
                    false, aF[i], false, bF[j], (short)0, acc[i][j], false, false);

        if (more) {   // commit B for next tile, then close the async fill
#pragma unroll
            for (int e = 0; e < 8; ++e) Bs[nxt][(bKp + e) * BST + bN]      = nb0[e];
#pragma unroll
            for (int e = 0; e < 8; ++e) Bs[nxt][(bKp + e) * BST + bN + 64] = nb1[e];
            wait_async();
        }
        __syncthreads();
    }

    // C/D layout: lane16 = N; VGPR v -> M = v + laneHi*8
#pragma unroll
    for (int i = 0; i < 2; ++i) {
#pragma unroll
        for (int j = 0; j < 4; ++j) {
            int gcol = blockN + waveN + j * 16 + lane16;
            float bv = bias[gcol];
            long long rbase = blockM + waveM + i * 16 + laneHi * 8;
#pragma unroll
            for (int v = 0; v < 8; ++v) {
                float val = acc[i][j][v] + bv;
                if (F16OUT)
                    ((half_t*)Y)[(rbase + v) * Nfull + gcol] = (half_t)val;
                else
                    ((float*)Y)[(rbase + v) * Nfull + gcol] = val;
            }
        }
    }
}

// ---------------------------------------------------------------------------
// Per-row MHA with S<=3 context rows. 1 block/row, 1 wave/head (wave32).
// ---------------------------------------------------------------------------
__global__ __launch_bounds__(256) void attn_kernel(
    const half_t* __restrict__ Q,
    const half_t* __restrict__ K0, const half_t* __restrict__ K1,
    const half_t* __restrict__ K2,
    const half_t* __restrict__ V0, const half_t* __restrict__ V1,
    const half_t* __restrict__ V2,
    half_t* __restrict__ outH, float* __restrict__ attnMean, int S)
{
    __shared__ float hsum[3];
    const int b = blockIdx.x;
    const int tid = threadIdx.x;
    const int lane = tid & 31;
    const long long off = (long long)b * Dm + (tid >> 5) * DH + lane * 4;

    if (tid < 3) hsum[tid] = 0.f;
    __syncthreads();

    v4h q4 = *(const v4h*)(Q + off);
    float q0 = (float)q4[0], q1 = (float)q4[1], q2 = (float)q4[2], q3 = (float)q4[3];

    const half_t* Ks[3] = {K0, K1, K2};
    const half_t* Vs[3] = {V0, V1, V2};

    float sc[3];
    for (int s = 0; s < S; ++s) {
        v4h k4 = *(const v4h*)(Ks[s] + off);
        float d = q0 * (float)k4[0] + q1 * (float)k4[1]
                + q2 * (float)k4[2] + q3 * (float)k4[3];
        for (int o = 16; o > 0; o >>= 1) d += __shfl_xor(d, o, 32);
        sc[s] = d * 0.08838834764831845f;   // 1/sqrt(128)
    }
    float m = sc[0];
    for (int s = 1; s < S; ++s) m = fmaxf(m, sc[s]);
    float a[3], den = 0.f;
    for (int s = 0; s < S; ++s) { a[s] = __expf(sc[s] - m); den += a[s]; }
    float inv = 1.f / den;

    float o0 = 0.f, o1 = 0.f, o2 = 0.f, o3 = 0.f;
    for (int s = 0; s < S; ++s) {
        a[s] *= inv;
        v4h v4 = *(const v4h*)(Vs[s] + off);
        o0 += a[s] * (float)v4[0]; o1 += a[s] * (float)v4[1];
        o2 += a[s] * (float)v4[2]; o3 += a[s] * (float)v4[3];
    }
    v4h r; r[0] = (half_t)o0; r[1] = (half_t)o1; r[2] = (half_t)o2; r[3] = (half_t)o3;
    *(v4h*)(outH + off) = r;

    if (lane == 0)
        for (int s = 0; s < S; ++s) atomicAdd(&hsum[s], a[s]);   // ds_add_f32
    __syncthreads();
    if (tid < S) attnMean[(long long)b * S + tid] = hsum[tid] * 0.125f;
}

// ---------------------------------------------------------------------------
// fused = LN((f + mag) + proj); writes fp32 out and optional f16 copy
// ---------------------------------------------------------------------------
__global__ __launch_bounds__(256) void ln_kernel(
    const float* __restrict__ f, const float* __restrict__ mag,
    const float* __restrict__ proj,
    const float* __restrict__ w, const float* __restrict__ bb,
    float* __restrict__ outF, half_t* __restrict__ outH)
{
    __shared__ float wsum[8], wsq[8];
    const int b = blockIdx.x;
    const int tid = threadIdx.x;
    const long long base = (long long)b * Dm;
    const int d0 = tid * 4;

    float x[4], s = 0.f, sq = 0.f;
#pragma unroll
    for (int i = 0; i < 4; ++i) {
        x[i] = f[base + d0 + i] + mag[d0 + i] + proj[base + d0 + i];
        s += x[i]; sq += x[i] * x[i];
    }
    for (int o = 16; o > 0; o >>= 1) {
        s  += __shfl_xor(s,  o, 32);
        sq += __shfl_xor(sq, o, 32);
    }
    if ((tid & 31) == 0) { wsum[tid >> 5] = s; wsq[tid >> 5] = sq; }
    __syncthreads();
    float ts = 0.f, tq = 0.f;
#pragma unroll
    for (int i = 0; i < 8; ++i) { ts += wsum[i]; tq += wsq[i]; }
    float mean = ts * (1.0f / Dm);
    float var  = tq * (1.0f / Dm) - mean * mean;
    float rstd = rsqrtf(var + 1e-5f);
#pragma unroll
    for (int i = 0; i < 4; ++i) {
        float y = (x[i] - mean) * rstd * w[d0 + i] + bb[d0 + i];
        outF[base + d0 + i] = y;
        if (outH) outH[base + d0 + i] = (half_t)y;
    }
}

// ---------------------------------------------------------------------------
extern "C" void kernel_launch(void* const* d_in, const int* in_sizes, int n_in,
                              void* d_out, int out_size, void* d_ws, size_t ws_size,
                              hipStream_t stream)
{
    (void)in_sizes; (void)n_in; (void)out_size; (void)ws_size;

    const float* f40  = (const float*)d_in[0];
    const float* f100 = (const float*)d_in[1];
    const float* f200 = (const float*)d_in[2];
    const float* f400 = (const float*)d_in[3];
    const float* mag  = (const float*)d_in[4];
    const float* in_w[3]  = {(const float*)d_in[5],  (const float*)d_in[11], (const float*)d_in[17]};
    const float* in_b[3]  = {(const float*)d_in[6],  (const float*)d_in[12], (const float*)d_in[18]};
    const float* out_w[3] = {(const float*)d_in[7],  (const float*)d_in[13], (const float*)d_in[19]};
    const float* out_b[3] = {(const float*)d_in[8],  (const float*)d_in[14], (const float*)d_in[20]};
    const float* ln_w[3]  = {(const float*)d_in[9],  (const float*)d_in[15], (const float*)d_in[21]};
    const float* ln_b[3]  = {(const float*)d_in[10], (const float*)d_in[16], (const float*)d_in[22]};

    float* out = (float*)d_out;
    float* hier[4]  = {out, out + BD, out + 2 * BD, out + 3 * BD};
    float* attnp[3];
    attnp[0] = out + 4 * BD;
    attnp[1] = attnp[0] + Bsz;
    attnp[2] = attnp[1] + 2 * Bsz;

    char* ws = (char*)d_ws;
    half_t* hbuf = (half_t*)ws;
    half_t* enhH[4]   = {hbuf, hbuf + BD, hbuf + 2 * BD, hbuf + 3 * BD};
    half_t* fusedH[2] = {hbuf + 4 * BD, hbuf + 5 * BD};
    half_t* qH        =  hbuf + 6 * BD;
    half_t* kH[3]     = {hbuf + 7 * BD,  hbuf + 8 * BD,  hbuf + 9 * BD};
    half_t* vH[3]     = {hbuf + 10 * BD, hbuf + 11 * BD, hbuf + 12 * BD};
    half_t* aoH       =  hbuf + 13 * BD;
    float*  projF     = (float*)(ws + 14 * BD * sizeof(half_t));
    half_t* wHbase    = (half_t*)(ws + 14 * BD * sizeof(half_t) + BD * sizeof(float));
    half_t* inwH[3], *outwH[3];
    for (int l = 0; l < 3; ++l) {
        inwH[l]  = wHbase + (long long)l * 4 * Dm * Dm;
        outwH[l] = inwH[l] + 3LL * Dm * Dm;
    }

    // 1) enh_i (f16) + hier[0] (fp32)
    prep_kernel<<<dim3((unsigned)((4 * BD) / 256)), 256, 0, stream>>>(
        f40, f100, f200, f400, mag,
        enhH[0], enhH[1], enhH[2], enhH[3], hier[0]);

    // 2) weights -> f16
    for (int l = 0; l < 3; ++l) {
        f32_to_f16_kernel<<<(3 * Dm * Dm) / 256, 256, 0, stream>>>(in_w[l], inwH[l], 3 * Dm * Dm);
        f32_to_f16_kernel<<<(Dm * Dm) / 256, 256, 0, stream>>>(out_w[l], outwH[l], Dm * Dm);
    }

    dim3 gg(Dm / TN, Bsz / TM);   // (8, 128)
    const half_t* ctxH[3]  = {enhH[0], fusedH[0], fusedH[1]};
    const float*  feats[3] = {f100, f200, f400};
    half_t* fout[3] = {fusedH[0], fusedH[1], nullptr};

    for (int l = 0; l < 3; ++l) {
        const int S = l + 1;
        const half_t* wq = inwH[l];
        const half_t* wk = inwH[l] + (long long)Dm * Dm;
        const half_t* wv = inwH[l] + 2LL * Dm * Dm;

        gemm_kernel<true><<<gg, 256, 0, stream>>>(enhH[l + 1], wq, in_b[l], qH, Dm, Dm);
        for (int s = 0; s < S; ++s) {
            gemm_kernel<true><<<gg, 256, 0, stream>>>(ctxH[s], wk, in_b[l] + Dm,     kH[s], Dm, Dm);
            gemm_kernel<true><<<gg, 256, 0, stream>>>(ctxH[s], wv, in_b[l] + 2 * Dm, vH[s], Dm, Dm);
        }
        attn_kernel<<<Bsz, 256, 0, stream>>>(qH, kH[0], kH[1], kH[2],
                                             vH[0], vH[1], vH[2], aoH, attnp[l], S);
        gemm_kernel<false><<<gg, 256, 0, stream>>>(aoH, outwH[l], out_b[l], projF, Dm, Dm);
        ln_kernel<<<Bsz, 256, 0, stream>>>(feats[l], mag + (l + 1) * Dm, projF,
                                           ln_w[l], ln_b[l], hier[l + 1], fout[l]);
    }
}
